// Bottleneck_57406532878654
// MI455X (gfx1250) — compile-verified, matrix operand order checked by reading, not run on hardware
//
#include <hip/hip_runtime.h>

// ---------------------------------------------------------------------------
// Fused quantized bottleneck for MI455X (gfx1250, wave32, WMMA bf16).
//   conv1 1x1 512->128  + BN1 + ReLU + BFP  -> A1 (bf16, exact BFP values)
//   conv2 3x3 128->128  + BN2 + ReLU + BFP  -> A2 (bf16)
//   conv3 1x1 128->512  + BN3 + residual + ReLU + BFP -> out (fp32)
// Weights pre-quantized (per 32-in-channel-block scale) to bf16 in workspace.
// Each wave owns 32 pixels x 128 out-channels: two 16-row A tiles sharing
// every B fragment (2 WMMAs per B load -> latency hiding + 2x intensity).
// ---------------------------------------------------------------------------

typedef __attribute__((ext_vector_type(16))) __bf16 bf16x16;
typedef __attribute__((ext_vector_type(8)))  float  f32x8;
typedef __attribute__((ext_vector_type(4)))  unsigned int u32x4;

union FragAB {
    bf16x16 v;
    u32x4   q[2];
};

#define DEV __device__ __forceinline__

static constexpr int NB   = 64;          // batch
static constexpr int HW   = 784;         // 28*28
static constexpr int C1   = 512;         // in/out channels
static constexpr int CM   = 128;         // mid channels
static constexpr int NPIX = NB * HW;     // 50176 = 392 * 128

// round-to-nearest-even float -> bf16 bits
DEV unsigned short f2bf(float f) {
    unsigned int u = __float_as_uint(f);
    unsigned int r = u + 0x7FFFu + ((u >> 16) & 1u);
    return (unsigned short)(r >> 16);
}

DEV f32x8 wmma_bf16(const FragAB& a, const FragAB& b, f32x8 c) {
    return __builtin_amdgcn_wmma_f32_16x16x32_bf16(
        /*neg_a=*/false, a.v, /*neg_b=*/false, b.v,
        /*c_mod=*/(short)0, c, /*reuse_a=*/false, /*reuse_b=*/false);
}

// Block-floating-point quantization of one 32-channel block (two 16-col tiles).
// C/D layout: VGPR r holds pixel M=r (lanes 0-15) / M=r+8 (lanes 16-31);
// a pixel's 32 channel values live in one VGPR slot across a 16-lane half of
// both tiles, so xor-reductions with masks <16 stay within the half.
DEV void bfp_block(f32x8& t0, f32x8& t1) {
#pragma unroll
    for (int r = 0; r < 8; ++r) {
        float m = fmaxf(fabsf(t0[r]), fabsf(t1[r]));
        m = fmaxf(m, __shfl_xor(m, 1, 32));
        m = fmaxf(m, __shfl_xor(m, 2, 32));
        m = fmaxf(m, __shfl_xor(m, 4, 32));
        m = fmaxf(m, __shfl_xor(m, 8, 32));
        m = fmaxf(m, 1e-24f);
        int ex = __builtin_amdgcn_frexp_expf(m);         // m = f*2^ex, f in [.5,1)
        float delta = ldexpf(1.0f, ex - 7);              // 2^(floor(log2 m) - 6)
        float rd = 1.0f / delta;
        t0[r] = fminf(fmaxf(rintf(t0[r] * rd), -128.0f), 127.0f) * delta;
        t1[r] = fminf(fmaxf(rintf(t1[r] * rd), -128.0f), 127.0f) * delta;
    }
}

// ---------------------------------------------------------------------------
// Weight quantization: per (O, in-block-of-32, kh, kw) scale = maxabs/127,
// q = round(w/alpha)*alpha, stored bf16 in [O][KHW][I] (B-matrix friendly:
// column O has contiguous K).
// ---------------------------------------------------------------------------
__global__ void __launch_bounds__(256)
wquant_kernel(const float* __restrict__ w, unsigned short* __restrict__ wq,
              int O, int I, int KHW) {
    int gid  = blockIdx.x * (blockDim.x >> 5) + (threadIdx.x >> 5);
    int lane = threadIdx.x & 31;
    int nblk = I >> 5;
    int total = O * KHW * nblk;
    if (gid >= total) return;
    int ib = gid % nblk;
    int t  = (gid / nblk) % KHW;
    int o  = gid / (nblk * KHW);
    int i  = ib * 32 + lane;
    float v = w[((size_t)o * I + i) * KHW + t];
    float a = fabsf(v);
#pragma unroll
    for (int msk = 16; msk; msk >>= 1) a = fmaxf(a, __shfl_xor(a, msk, 32));
    float alpha = fmaxf(a * (1.0f / 127.0f), 1e-24f);
    float q = rintf(v / alpha) * alpha;
    wq[((size_t)o * KHW + t) * I + i] = f2bf(q);
}

// ---------------------------------------------------------------------------
// conv1: GEMM  M=NPIX (pixels), N=128, K=512.  x is NCHW fp32: A(p,c) strided
// by HW, so stage two 16px x 32ch tiles through LDS (coalesced along pixels).
// Fused BN1+ReLU+BFP -> A1 bf16 [p][c].
// ---------------------------------------------------------------------------
__global__ void __launch_bounds__(128)
conv1_kernel(const float* __restrict__ x, const unsigned short* __restrict__ w1q,
             const float* __restrict__ bng, const float* __restrict__ bnb,
             const float* __restrict__ bnm, const float* __restrict__ bnv,
             unsigned short* __restrict__ A1) {
    __shared__ unsigned short a_lds[4][2 * 16 * 32];
    __shared__ unsigned short o_lds[4][16 * CM];
    const int wv = threadIdx.x >> 5, lane = threadIdx.x & 31;
    const int pg = (blockIdx.x * 4 + wv) * 2;    // two consecutive 16-px groups
    int p0t[2], nt[2], hw0t[2];
    const float* xbt[2];
#pragma unroll
    for (int s = 0; s < 2; ++s) {
        p0t[s] = (pg + s) * 16;
        nt[s]  = p0t[s] / HW;                    // 16 | HW: tile stays in image
        hw0t[s] = p0t[s] - nt[s] * HW;
        xbt[s] = x + (size_t)nt[s] * C1 * HW + hw0t[s];
    }

    const f32x8 zero8 = {0.f, 0.f, 0.f, 0.f, 0.f, 0.f, 0.f, 0.f};
    f32x8 acc[2][8];
#pragma unroll
    for (int s = 0; s < 2; ++s)
#pragma unroll
        for (int t = 0; t < 8; ++t) acc[s][t] = zero8;

    const int m  = lane & 15;
    const int k0 = (lane >> 4) * 8;              // A/B 16-bit fragment K split
    const int px = lane & 15, cof = lane >> 4;

    for (int kb = 0; kb < C1 / 32; ++kb) {
        // stage 2 x (16px x 32ch), fp32 -> bf16, coalesced along pixels
#pragma unroll
        for (int s = 0; s < 2; ++s)
#pragma unroll
            for (int i = 0; i < 16; ++i) {
                int c = kb * 32 + i * 2 + cof;
                a_lds[wv][s * 512 + px * 32 + i * 2 + cof] =
                    f2bf(xbt[s][(size_t)c * HW + px]);
            }
        __syncthreads();
        FragAB a0, a1;
        a0.q[0] = *(const u32x4*)&a_lds[wv][m * 32 + k0];
        a0.q[1] = *(const u32x4*)&a_lds[wv][m * 32 + k0 + 16];
        a1.q[0] = *(const u32x4*)&a_lds[wv][512 + m * 32 + k0];
        a1.q[1] = *(const u32x4*)&a_lds[wv][512 + m * 32 + k0 + 16];
        const unsigned short* wp = w1q + kb * 32 + k0;
#pragma unroll
        for (int t = 0; t < 8; ++t) {
            int o = t * 16 + m;
            FragAB b;
            b.q[0] = *(const u32x4*)(wp + (size_t)o * C1);
            b.q[1] = *(const u32x4*)(wp + (size_t)o * C1 + 16);
            acc[0][t] = wmma_bf16(a0, b, acc[0][t]);   // 2 WMMAs per B load
            acc[1][t] = wmma_bf16(a1, b, acc[1][t]);
        }
        __syncthreads();
    }
    // folded BN coefficients (shared by both tiles)
    float scv[8], biv[8];
#pragma unroll
    for (int t = 0; t < 8; ++t) {
        int c = t * 16 + m;
        scv[t] = bng[c] * rsqrtf(bnv[c] + 1e-5f);
        biv[t] = bnb[c] - bnm[c] * scv[t];
    }
    const int rowhi = (lane >> 4) * 8;
#pragma unroll
    for (int s = 0; s < 2; ++s) {
#pragma unroll
        for (int t = 0; t < 8; ++t)
#pragma unroll
            for (int r = 0; r < 8; ++r)
                acc[s][t][r] = fmaxf(acc[s][t][r] * scv[t] + biv[t], 0.0f);
#pragma unroll
        for (int b = 0; b < 4; ++b) bfp_block(acc[s][2 * b], acc[s][2 * b + 1]);
        // LDS transpose -> coalesced b128 bf16 stores, layout A1[p][c]
#pragma unroll
        for (int t = 0; t < 8; ++t) {
            int c = t * 16 + m;
#pragma unroll
            for (int r = 0; r < 8; ++r)
                o_lds[wv][(r + rowhi) * CM + c] = f2bf(acc[s][t][r]);
        }
        __syncthreads();
        unsigned short* A1p = A1 + (size_t)p0t[s] * CM;
#pragma unroll
        for (int i = 0; i < 8; ++i) {
            int chunk = i * 32 + lane;
            int rr = chunk >> 4, cc = (chunk & 15) * 8;
            *(u32x4*)(A1p + (size_t)rr * CM + cc) =
                *(const u32x4*)&o_lds[wv][rr * CM + cc];
        }
        __syncthreads();
    }
}

// ---------------------------------------------------------------------------
// conv2: implicit GEMM, 3x3 pad=1.  K = 9 taps x 4 blocks of 32.  A1 is bf16
// channel-contiguous so each lane b128-loads its row fragment directly;
// out-of-image taps zero-fill.  Fused BN2+ReLU+BFP -> A2 bf16.
// ---------------------------------------------------------------------------
__global__ void __launch_bounds__(128)
conv2_kernel(const unsigned short* __restrict__ A1, const unsigned short* __restrict__ w2q,
             const float* __restrict__ bng, const float* __restrict__ bnb,
             const float* __restrict__ bnm, const float* __restrict__ bnv,
             unsigned short* __restrict__ A2) {
    __shared__ unsigned short o_lds[4][16 * CM];
    const int wv = threadIdx.x >> 5, lane = threadIdx.x & 31;
    const int pg = (blockIdx.x * 4 + wv) * 2;
    const int m  = lane & 15;
    const int k0 = (lane >> 4) * 8;

    int p0t[2], nt[2], hw0t[2], ht[2], wt[2];
#pragma unroll
    for (int s = 0; s < 2; ++s) {
        p0t[s] = (pg + s) * 16;
        nt[s]  = p0t[s] / HW;
        hw0t[s] = p0t[s] - nt[s] * HW;
        int hwm = hw0t[s] + m;
        ht[s] = hwm / 28;
        wt[s] = hwm % 28;
    }

    const f32x8 zero8 = {0.f, 0.f, 0.f, 0.f, 0.f, 0.f, 0.f, 0.f};
    const u32x4 zq = {0u, 0u, 0u, 0u};
    f32x8 acc[2][8];
#pragma unroll
    for (int s = 0; s < 2; ++s)
#pragma unroll
        for (int t = 0; t < 8; ++t) acc[s][t] = zero8;

    for (int tap = 0; tap < 9; ++tap) {
        const int dy = tap / 3 - 1, dx = tap % 3 - 1;
        bool okt[2];
        const unsigned short* apt[2];
#pragma unroll
        for (int s = 0; s < 2; ++s) {
            int hh = ht[s] + dy, ww = wt[s] + dx;
            okt[s] = ((unsigned)hh < 28u) && ((unsigned)ww < 28u);
            size_t pidx = okt[s] ? (size_t)(nt[s] * HW + hh * 28 + ww) : 0;
            apt[s] = A1 + pidx * CM + k0;
            __builtin_prefetch(apt[s], 0, 0);              // global_prefetch_b8
        }
        const unsigned short* wp = w2q + tap * CM + k0;
        for (int kb = 0; kb < 4; ++kb) {
            FragAB a0, a1;
            if (okt[0]) {
                a0.q[0] = *(const u32x4*)(apt[0] + kb * 32);
                a0.q[1] = *(const u32x4*)(apt[0] + kb * 32 + 16);
            } else { a0.q[0] = zq; a0.q[1] = zq; }
            if (okt[1]) {
                a1.q[0] = *(const u32x4*)(apt[1] + kb * 32);
                a1.q[1] = *(const u32x4*)(apt[1] + kb * 32 + 16);
            } else { a1.q[0] = zq; a1.q[1] = zq; }
#pragma unroll
            for (int t = 0; t < 8; ++t) {
                int o = t * 16 + m;
                FragAB b;
                b.q[0] = *(const u32x4*)(wp + (size_t)o * (9 * CM) + kb * 32);
                b.q[1] = *(const u32x4*)(wp + (size_t)o * (9 * CM) + kb * 32 + 16);
                acc[0][t] = wmma_bf16(a0, b, acc[0][t]);
                acc[1][t] = wmma_bf16(a1, b, acc[1][t]);
            }
        }
    }
    float scv[8], biv[8];
#pragma unroll
    for (int t = 0; t < 8; ++t) {
        int c = t * 16 + m;
        scv[t] = bng[c] * rsqrtf(bnv[c] + 1e-5f);
        biv[t] = bnb[c] - bnm[c] * scv[t];
    }
    const int rowhi = (lane >> 4) * 8;
#pragma unroll
    for (int s = 0; s < 2; ++s) {
#pragma unroll
        for (int t = 0; t < 8; ++t)
#pragma unroll
            for (int r = 0; r < 8; ++r)
                acc[s][t][r] = fmaxf(acc[s][t][r] * scv[t] + biv[t], 0.0f);
#pragma unroll
        for (int b = 0; b < 4; ++b) bfp_block(acc[s][2 * b], acc[s][2 * b + 1]);
#pragma unroll
        for (int t = 0; t < 8; ++t) {
            int c = t * 16 + m;
#pragma unroll
            for (int r = 0; r < 8; ++r)
                o_lds[wv][(r + rowhi) * CM + c] = f2bf(acc[s][t][r]);
        }
        __syncthreads();
        unsigned short* A2p = A2 + (size_t)p0t[s] * CM;
#pragma unroll
        for (int i = 0; i < 8; ++i) {
            int chunk = i * 32 + lane;
            int rr = chunk >> 4, cc = (chunk & 15) * 8;
            *(u32x4*)(A2p + (size_t)rr * CM + cc) =
                *(const u32x4*)&o_lds[wv][rr * CM + cc];
        }
        __syncthreads();
    }
}

// ---------------------------------------------------------------------------
// conv3: GEMM M=NPIX, N=512 (split in 4 channel quarters over grid.y), K=128.
// Fused BN3 + residual add + ReLU + BFP -> out fp32 NCHW (LDS transpose for
// coalesced b128 stores along HW).
// ---------------------------------------------------------------------------
__global__ void __launch_bounds__(128)
conv3_kernel(const unsigned short* __restrict__ A2, const unsigned short* __restrict__ w3q,
             const float* __restrict__ bng, const float* __restrict__ bnb,
             const float* __restrict__ bnm, const float* __restrict__ bnv,
             const float* __restrict__ xin, float* __restrict__ out) {
    __shared__ float o_lds[4][CM * 16];               // [c][px] per wave
    const int wv = threadIdx.x >> 5, lane = threadIdx.x & 31;
    const int pg = (blockIdx.x * 4 + wv) * 2;
    const int cq = blockIdx.y * CM;                   // channel quarter base
    const int m  = lane & 15;
    const int k0 = (lane >> 4) * 8;

    int p0t[2], nt[2], hw0t[2];
#pragma unroll
    for (int s = 0; s < 2; ++s) {
        p0t[s] = (pg + s) * 16;
        nt[s]  = p0t[s] / HW;
        hw0t[s] = p0t[s] - nt[s] * HW;
    }

    const f32x8 zero8 = {0.f, 0.f, 0.f, 0.f, 0.f, 0.f, 0.f, 0.f};
    f32x8 acc[2][8];
#pragma unroll
    for (int s = 0; s < 2; ++s)
#pragma unroll
        for (int t = 0; t < 8; ++t) acc[s][t] = zero8;

    const unsigned short* ap0 = A2 + (size_t)(p0t[0] + m) * CM + k0;
    const unsigned short* ap1 = A2 + (size_t)(p0t[1] + m) * CM + k0;
    for (int kb = 0; kb < 4; ++kb) {
        FragAB a0, a1;
        a0.q[0] = *(const u32x4*)(ap0 + kb * 32);
        a0.q[1] = *(const u32x4*)(ap0 + kb * 32 + 16);
        a1.q[0] = *(const u32x4*)(ap1 + kb * 32);
        a1.q[1] = *(const u32x4*)(ap1 + kb * 32 + 16);
#pragma unroll
        for (int t = 0; t < 8; ++t) {
            int o = cq + t * 16 + m;
            FragAB b;
            b.q[0] = *(const u32x4*)(w3q + (size_t)o * CM + kb * 32 + k0);
            b.q[1] = *(const u32x4*)(w3q + (size_t)o * CM + kb * 32 + k0 + 16);
            acc[0][t] = wmma_bf16(a0, b, acc[0][t]);
            acc[1][t] = wmma_bf16(a1, b, acc[1][t]);
        }
    }
    float scv[8], biv[8];
#pragma unroll
    for (int t = 0; t < 8; ++t) {
        int c = cq + t * 16 + m;
        scv[t] = bng[c] * rsqrtf(bnv[c] + 1e-5f);
        biv[t] = bnb[c] - bnm[c] * scv[t];
    }
    const int rowhi = (lane >> 4) * 8;
#pragma unroll
    for (int s = 0; s < 2; ++s) {
#pragma unroll
        for (int t = 0; t < 8; ++t) {
            int c = cq + t * 16 + m;
            const float* xp = xin + ((size_t)nt[s] * C1 + c) * HW + hw0t[s] + rowhi;
#pragma unroll
            for (int r = 0; r < 8; ++r)
                acc[s][t][r] = fmaxf(acc[s][t][r] * scv[t] + biv[t] + xp[r], 0.0f);
        }
#pragma unroll
        for (int b = 0; b < 4; ++b) bfp_block(acc[s][2 * b], acc[s][2 * b + 1]);
        // transpose [c][px] then coalesced fp32 b128 stores along HW
#pragma unroll
        for (int t = 0; t < 8; ++t) {
            int cl = t * 16 + m;
#pragma unroll
            for (int r = 0; r < 8; ++r) o_lds[wv][cl * 16 + r + rowhi] = acc[s][t][r];
        }
        __syncthreads();
        float* op = out + ((size_t)nt[s] * C1 + cq) * HW + hw0t[s];
#pragma unroll
        for (int i = 0; i < 16; ++i) {
            int chunk = i * 32 + lane;                // 512 chunks of 4 floats
            int c = chunk >> 2, pxq = (chunk & 3) * 4;
            *(u32x4*)(op + (size_t)c * HW + pxq) =
                *(const u32x4*)&o_lds[wv][c * 16 + pxq];
        }
        __syncthreads();
    }
}

// ---------------------------------------------------------------------------
extern "C" void kernel_launch(void* const* d_in, const int* in_sizes, int n_in,
                              void* d_out, int out_size, void* d_ws, size_t ws_size,
                              hipStream_t stream) {
    (void)in_sizes; (void)n_in; (void)out_size; (void)ws_size;
    const float* x   = (const float*)d_in[0];
    const float* w1  = (const float*)d_in[1];
    const float* w2  = (const float*)d_in[2];
    const float* w3  = (const float*)d_in[3];
    const float* bn1g = (const float*)d_in[4],  *bn1b = (const float*)d_in[5];
    const float* bn1m = (const float*)d_in[6],  *bn1v = (const float*)d_in[7];
    const float* bn2g = (const float*)d_in[8],  *bn2b = (const float*)d_in[9];
    const float* bn2m = (const float*)d_in[10], *bn2v = (const float*)d_in[11];
    const float* bn3g = (const float*)d_in[12], *bn3b = (const float*)d_in[13];
    const float* bn3m = (const float*)d_in[14], *bn3v = (const float*)d_in[15];

    // workspace layout (bf16 = unsigned short), ~26.2 MB total
    unsigned short* w1q = (unsigned short*)d_ws;                 // 128*512
    unsigned short* w2q = w1q + 128 * 512;                       // 128*9*128
    unsigned short* w3q = w2q + 128 * 9 * 128;                   // 512*128
    unsigned short* A1  = w3q + 512 * 128;                       // NPIX*128
    unsigned short* A2  = A1 + (size_t)NPIX * CM;                // NPIX*128

    // weight quantization: one wave per (O, khw, 32-in-block) group
    wquant_kernel<<<(128 * 1 * 16 + 7) / 8, 256, 0, stream>>>(w1, w1q, 128, 512, 1);
    wquant_kernel<<<(128 * 9 * 4 + 7) / 8, 256, 0, stream>>>(w2, w2q, 128, 128, 9);
    wquant_kernel<<<(512 * 1 * 4 + 7) / 8, 256, 0, stream>>>(w3, w3q, 512, 128, 1);

    // 4 waves/WG, each wave = 32 px x 128 ch  ->  128 px per WG
    conv1_kernel<<<NPIX / 128, 128, 0, stream>>>(x, w1q, bn1g, bn1b, bn1m, bn1v, A1);
    conv2_kernel<<<NPIX / 128, 128, 0, stream>>>(A1, w2q, bn2g, bn2b, bn2m, bn2v, A2);
    dim3 g3(NPIX / 128, 4);
    conv3_kernel<<<g3, 128, 0, stream>>>(A2, w3q, bn3g, bn3b, bn3m, bn3v, x, (float*)d_out);
}
